// GAT_59992103190782
// MI455X (gfx1250) — matrix operand order repaired and checked by reading, not compiled, not run on hardware
//
#include <hip/hip_runtime.h>
#include <hip/hip_bf16.h>

// ---------------------------------------------------------------------------
// GAT forward for MI455X (gfx1250, wave32).
// K=128 GEMMs via v_wmma_f32_16x16x32_f16 (unconditional b128 loads, B held in
// registers across M-tiles); edge softmax/aggregation via f32 atomics.
// ---------------------------------------------------------------------------

typedef __attribute__((ext_vector_type(16))) _Float16 v16h;
typedef __attribute__((ext_vector_type(8)))  float    v8f;

#define N_NODES 50000
#define N_EDGES 800000
#define E_TOT   (N_EDGES + N_NODES)   // with self loops
#define N_GRAPHS 512
#define HID 128

// -------- order-preserving float<->uint for atomic max --------
__device__ __forceinline__ unsigned fenc(float f) {
    unsigned u = __float_as_uint(f);
    return (u & 0x80000000u) ? ~u : (u | 0x80000000u);
}
__device__ __forceinline__ float fdec(unsigned u) {
    unsigned v = (u & 0x80000000u) ? (u & 0x7FFFFFFFu) : ~u;
    return __uint_as_float(v);
}
#define ENC_NEG_INF 0x007FFFFFu   // fenc(-inf)

// ---------------------------------------------------------------------------
// Generic fill kernels
// ---------------------------------------------------------------------------
__global__ void fill_f32(float* p, float v, int n) {
    int i = blockIdx.x * blockDim.x + threadIdx.x;
    if (i < n) p[i] = v;
}
__global__ void fill_u32(unsigned* p, unsigned v, int n) {
    int i = blockIdx.x * blockDim.x + threadIdx.x;
    if (i < n) p[i] = v;
}

// ---------------------------------------------------------------------------
// Layer-0 GEMM (K=7, tiny): H[n, c..c+3] = sum_k X[n,k] * W[k, c..c+3]
// One thread per (node, 4 channels).
// ---------------------------------------------------------------------------
__global__ void gemm_in7(const float* __restrict__ X, const float* __restrict__ W,
                         float* __restrict__ H) {
    int t = blockIdx.x * blockDim.x + threadIdx.x;
    if (t >= N_NODES * 32) return;
    int n  = t >> 5;
    int c4 = (t & 31) * 4;
    const float* xr = X + (size_t)n * 7;
    float4 acc = make_float4(0.f, 0.f, 0.f, 0.f);
#pragma unroll
    for (int k = 0; k < 7; ++k) {
        float xk = xr[k];
        const float4 w = *(const float4*)(W + (size_t)k * 128 + c4);
        acc.x = fmaf(xk, w.x, acc.x);
        acc.y = fmaf(xk, w.y, acc.y);
        acc.z = fmaf(xk, w.z, acc.z);
        acc.w = fmaf(xk, w.w, acc.w);
    }
    *(float4*)(H + (size_t)n * 128 + c4) = acc;
}

// ---------------------------------------------------------------------------
// WMMA GEMM, K = 128 exactly, M a multiple of 16 (no guards anywhere):
//   Hout[M,128] = X[M,128] @ W[128,128]
// 8 waves/block, one wave per 16-column stripe. Each wave preloads its whole
// 128x16 B stripe as 4 f16 fragments (registers), then sweeps mtPerBlock
// M-tiles: per k-step, A fragment = 4 unconditional float4 loads + cvt.
// mode: 0 = raw, 1 = +bias then ReLU
// ---------------------------------------------------------------------------
__device__ __forceinline__ v16h load_a_frag(const float* __restrict__ xrow,
                                            int k0, int hf) {
    const float4 p0 = *(const float4*)(xrow + k0 + hf * 8);
    const float4 p1 = *(const float4*)(xrow + k0 + hf * 8 + 4);
    const float4 p2 = *(const float4*)(xrow + k0 + 16 + hf * 8);
    const float4 p3 = *(const float4*)(xrow + k0 + 16 + hf * 8 + 4);
    v16h a;
    a[0]  = (_Float16)p0.x; a[1]  = (_Float16)p0.y; a[2]  = (_Float16)p0.z; a[3]  = (_Float16)p0.w;
    a[4]  = (_Float16)p1.x; a[5]  = (_Float16)p1.y; a[6]  = (_Float16)p1.z; a[7]  = (_Float16)p1.w;
    a[8]  = (_Float16)p2.x; a[9]  = (_Float16)p2.y; a[10] = (_Float16)p2.z; a[11] = (_Float16)p2.w;
    a[12] = (_Float16)p3.x; a[13] = (_Float16)p3.y; a[14] = (_Float16)p3.z; a[15] = (_Float16)p3.w;
    return a;
}

__global__ void gemm128_wmma(const float* __restrict__ X, const float* __restrict__ W,
                             const float* __restrict__ bias, float* __restrict__ Hout,
                             int Mtiles, int mtPerBlock, int mode) {
    const int wave = threadIdx.x >> 5;       // 0..7  -> 16-col stripe
    const int lane = threadIdx.x & 31;
    const int n0   = wave * 16;
    const int r    = lane & 15;
    const int hf   = lane >> 4;

    // Preload B fragments for all 4 k-steps (K=128), f32 -> f16.
    // Lane r holds column (n0+r); fragment element e holds row K = ks*32 + (e<8 ? hf*8+e : 16+hf*8+e-8).
    v16h b0, b1, b2, b3;
#pragma unroll
    for (int e = 0; e < 16; ++e) {
        const int krow = (e < 8) ? (hf * 8 + e) : (16 + hf * 8 + (e - 8));
        b0[e] = (_Float16)W[(size_t)(krow)       * 128 + n0 + r];
        b1[e] = (_Float16)W[(size_t)(krow + 32)  * 128 + n0 + r];
        b2[e] = (_Float16)W[(size_t)(krow + 64)  * 128 + n0 + r];
        b3[e] = (_Float16)W[(size_t)(krow + 96)  * 128 + n0 + r];
    }

    for (int mt = 0; mt < mtPerBlock; ++mt) {
        const int mtile = blockIdx.x * mtPerBlock + mt;
        if (mtile >= Mtiles) break;          // uniform; never hit with exact grids
        const float* xrow = X + (size_t)(mtile * 16 + r) * 128;

        v8f c = {};
        v16h a;
        a = load_a_frag(xrow,  0, hf);
        c = __builtin_amdgcn_wmma_f32_16x16x32_f16(false, a, false, b0, (short)0, c, false, false);
        a = load_a_frag(xrow, 32, hf);
        c = __builtin_amdgcn_wmma_f32_16x16x32_f16(false, a, false, b1, (short)0, c, false, false);
        a = load_a_frag(xrow, 64, hf);
        c = __builtin_amdgcn_wmma_f32_16x16x32_f16(false, a, false, b2, (short)0, c, false, false);
        a = load_a_frag(xrow, 96, hf);
        c = __builtin_amdgcn_wmma_f32_16x16x32_f16(false, a, false, b3, (short)0, c, false, false);

        // C/D layout: VGPR rr -> row (8*hf + rr), lane r -> column (n0 + r)
        const int col = n0 + r;
#pragma unroll
        for (int rr = 0; rr < 8; ++rr) {
            float v = c[rr];
            if (mode == 1) { v += bias[col]; v = fmaxf(v, 0.f); }
            Hout[(size_t)(mtile * 16 + hf * 8 + rr) * 128 + col] = v;
        }
    }
}

// ---------------------------------------------------------------------------
// Per-node attention logits: as[n] = h[n,:].a_src, ad[n] = h[n,:].a_dst
// ---------------------------------------------------------------------------
__global__ void alpha_kernel(const float* __restrict__ h,
                             const float* __restrict__ a_s,
                             const float* __restrict__ a_d,
                             float* __restrict__ as_out, float* __restrict__ ad_out) {
    int n = blockIdx.x * blockDim.x + threadIdx.x;
    if (n >= N_NODES) return;
    const float* row = h + (size_t)n * 128;
    float s1 = 0.f, s2 = 0.f;
#pragma unroll 8
    for (int c = 0; c < 128; ++c) {
        float v = row[c];
        s1 = fmaf(v, a_s[c], s1);
        s2 = fmaf(v, a_d[c], s2);
    }
    as_out[n] = s1;
    ad_out[n] = s2;
}

// ---------------------------------------------------------------------------
// Edge pass 1: e = leaky_relu(as[src] + ad[dst]); segment max over dst
// ---------------------------------------------------------------------------
__global__ void edge_max_kernel(const long long* __restrict__ ei,
                                const float* __restrict__ as_,
                                const float* __restrict__ ad_,
                                float* __restrict__ e_out,
                                unsigned* __restrict__ m_enc) {
    int i = blockIdx.x * blockDim.x + threadIdx.x;
    if (i >= E_TOT) return;
    int s, d;
    if (i < N_EDGES) { s = (int)ei[i]; d = (int)ei[N_EDGES + i]; }
    else             { s = d = i - N_EDGES; }
    float e = as_[s] + ad_[d];
    e = e > 0.f ? e : 0.2f * e;
    e_out[i] = e;
    atomicMax(m_enc + d, fenc(e));
}

// ---------------------------------------------------------------------------
// Edge pass 2: e_exp = exp(e - m[dst]); segment sum over dst
// ---------------------------------------------------------------------------
__global__ void edge_exp_kernel(const long long* __restrict__ ei,
                                const float* __restrict__ e_in,
                                const unsigned* __restrict__ m_enc,
                                float* __restrict__ eexp,
                                float* __restrict__ ssum) {
    int i = blockIdx.x * blockDim.x + threadIdx.x;
    if (i >= E_TOT) return;
    int d = (i < N_EDGES) ? (int)ei[N_EDGES + i] : (i - N_EDGES);
    float m = fdec(m_enc[d]);
    if (!isfinite(m)) m = 0.f;
    float ex = __expf(e_in[i] - m);
    eexp[i] = ex;
    atomicAdd(ssum + d, ex);
}

// ---------------------------------------------------------------------------
// Edge pass 3: out[dst,:] += (e_exp / s[dst]) * h[src,:]
// One wave32 per edge; each lane handles 4 channels (float4).
// ---------------------------------------------------------------------------
__global__ void edge_aggr_kernel(const long long* __restrict__ ei,
                                 const float* __restrict__ eexp,
                                 const float* __restrict__ ssum,
                                 const float* __restrict__ h,
                                 float* __restrict__ out) {
    int w = (int)((blockIdx.x * (size_t)blockDim.x + threadIdx.x) >> 5);
    int lane = threadIdx.x & 31;
    if (w >= E_TOT) return;
    int s, d;
    if (w < N_EDGES) { s = (int)ei[w]; d = (int)ei[N_EDGES + w]; }
    else             { s = d = w - N_EDGES; }
    float alpha = eexp[w] / (ssum[d] + 1e-16f);
    const float4* hs = (const float4*)(h + (size_t)s * 128);
    float4 v = hs[lane];
    float* o = out + (size_t)d * 128 + lane * 4;
    atomicAdd(o + 0, alpha * v.x);
    atomicAdd(o + 1, alpha * v.y);
    atomicAdd(o + 2, alpha * v.z);
    atomicAdd(o + 3, alpha * v.w);
}

// ---------------------------------------------------------------------------
// x_next = elu(agg + bias)   (in place over agg buffer)
// ---------------------------------------------------------------------------
__global__ void node_finish_kernel(float* __restrict__ agg,
                                   const float* __restrict__ bias) {
    int idx = blockIdx.x * blockDim.x + threadIdx.x;
    if (idx >= N_NODES * 128) return;
    float v = agg[idx] + bias[idx & 127];
    agg[idx] = v > 0.f ? v : (__expf(v) - 1.f);
}

// ---------------------------------------------------------------------------
// Mean-pool accumulation: one wave per node
// ---------------------------------------------------------------------------
__global__ void pool_acc_kernel(const float* __restrict__ x,
                                const long long* __restrict__ batch,
                                float* __restrict__ sums, float* __restrict__ cnt) {
    int w = (int)((blockIdx.x * (size_t)blockDim.x + threadIdx.x) >> 5);
    int lane = threadIdx.x & 31;
    if (w >= N_NODES) return;
    int g = (int)batch[w];
    const float4* xr = (const float4*)(x + (size_t)w * 128);
    float4 v = xr[lane];
    float* o = sums + (size_t)g * 128 + lane * 4;
    atomicAdd(o + 0, v.x);
    atomicAdd(o + 1, v.y);
    atomicAdd(o + 2, v.z);
    atomicAdd(o + 3, v.w);
    if (lane == 0) atomicAdd(cnt + g, 1.f);
}

__global__ void pool_mean_kernel(const float* __restrict__ sums,
                                 const float* __restrict__ cnt,
                                 float* __restrict__ g) {
    int idx = blockIdx.x * blockDim.x + threadIdx.x;
    if (idx >= N_GRAPHS * 128) return;
    float c = cnt[idx >> 7];
    g[idx] = sums[idx] / fmaxf(c, 1.f);
}

// ---------------------------------------------------------------------------
// Head: logits = h1 @ fc2_w + fc2_b ; log_softmax -> out[G,2]
// ---------------------------------------------------------------------------
__global__ void head_kernel(const float* __restrict__ h1,
                            const float* __restrict__ w2,
                            const float* __restrict__ b2,
                            float* __restrict__ out) {
    int g = blockIdx.x * blockDim.x + threadIdx.x;
    if (g >= N_GRAPHS) return;
    const float* r = h1 + (size_t)g * 128;
    float l0 = b2[0], l1 = b2[1];
#pragma unroll 8
    for (int k = 0; k < 128; ++k) {
        float v = r[k];
        l0 = fmaf(v, w2[k * 2 + 0], l0);
        l1 = fmaf(v, w2[k * 2 + 1], l1);
    }
    float m = fmaxf(l0, l1);
    float lse = m + __logf(__expf(l0 - m) + __expf(l1 - m));
    out[g * 2 + 0] = l0 - lse;
    out[g * 2 + 1] = l1 - lse;
}

// ---------------------------------------------------------------------------
// Launcher
// ---------------------------------------------------------------------------
extern "C" void kernel_launch(void* const* d_in, const int* in_sizes, int n_in,
                              void* d_out, int out_size, void* d_ws, size_t ws_size,
                              hipStream_t stream) {
    const float*     x_in  = (const float*)d_in[0];
    const long long* ei    = (const long long*)d_in[1];
    const long long* batch = (const long long*)d_in[2];
    const float* W[3]  = {(const float*)d_in[3],  (const float*)d_in[7],  (const float*)d_in[11]};
    const float* as[3] = {(const float*)d_in[4],  (const float*)d_in[8],  (const float*)d_in[12]};
    const float* ad[3] = {(const float*)d_in[5],  (const float*)d_in[9],  (const float*)d_in[13]};
    const float* bb[3] = {(const float*)d_in[6],  (const float*)d_in[10], (const float*)d_in[14]};
    const float* fc1_w = (const float*)d_in[15];
    const float* fc1_b = (const float*)d_in[16];
    const float* fc2_w = (const float*)d_in[17];
    const float* fc2_b = (const float*)d_in[18];
    float* out = (float*)d_out;

    // ---- workspace layout (floats) ----
    float*    ws    = (float*)d_ws;
    float*    bufA  = ws;                        // 6.4M  node features / agg
    float*    bufB  = bufA + (size_t)N_NODES * 128;
    float*    bufH  = bufB + (size_t)N_NODES * 128;  // h = X @ W
    float*    as_v  = bufH + (size_t)N_NODES * 128;  // 50000
    float*    ad_v  = as_v + N_NODES;
    unsigned* m_enc = (unsigned*)(ad_v + N_NODES);   // 50000
    float*    ssum  = (float*)(m_enc + N_NODES);     // 50000
    float*    e_val = ssum + N_NODES;                // 850000
    float*    e_exp = e_val + E_TOT;                 // 850000
    float*    sums  = e_exp + E_TOT;                 // 65536
    float*    cnt   = sums + (size_t)N_GRAPHS * 128; // 512
    float*    gbuf  = cnt + N_GRAPHS;                // 65536
    float*    h1    = gbuf + (size_t)N_GRAPHS * 128; // 65536

    const int T = 256;
    const int edge_blocks      = (E_TOT + T - 1) / T;
    const int wave_edge_blocks = (int)(((size_t)E_TOT * 32 + T - 1) / T);
    const int wave_node_blocks = (int)(((size_t)N_NODES * 32 + T - 1) / T);
    const int nh_blocks        = (N_NODES * 128 + T - 1) / T;

    const float* Xl[3]   = {x_in, bufA, bufB};
    float*       aggl[3] = {bufA, bufB, bufA};

    for (int l = 0; l < 3; ++l) {
        // h = X @ W
        if (l == 0) {
            gemm_in7<<<(N_NODES * 32 + T - 1) / T, T, 0, stream>>>(x_in, W[0], bufH);
        } else {
            // 3125 M-tiles = 625 blocks x 5 tiles, exact
            gemm128_wmma<<<625, T, 0, stream>>>(Xl[l], W[l], nullptr, bufH,
                                                N_NODES / 16, 5, 0);
        }
        // attention logits per node
        alpha_kernel<<<(N_NODES + T - 1) / T, T, 0, stream>>>(bufH, as[l], ad[l], as_v, ad_v);
        // reset segment buffers + aggregation target
        fill_u32<<<(N_NODES + T - 1) / T, T, 0, stream>>>(m_enc, ENC_NEG_INF, N_NODES);
        fill_f32<<<(N_NODES + T - 1) / T, T, 0, stream>>>(ssum, 0.f, N_NODES);
        fill_f32<<<nh_blocks, T, 0, stream>>>(aggl[l], 0.f, N_NODES * 128);
        // segment softmax + aggregation over edges
        edge_max_kernel<<<edge_blocks, T, 0, stream>>>(ei, as_v, ad_v, e_val, m_enc);
        edge_exp_kernel<<<edge_blocks, T, 0, stream>>>(ei, e_val, m_enc, e_exp, ssum);
        edge_aggr_kernel<<<wave_edge_blocks, T, 0, stream>>>(ei, e_exp, ssum, bufH, aggl[l]);
        // + bias, ELU
        node_finish_kernel<<<nh_blocks, T, 0, stream>>>(aggl[l], bb[l]);
    }

    // ---- global mean pool (final features live in bufA) ----
    fill_f32<<<(N_GRAPHS * 128 + T - 1) / T, T, 0, stream>>>(sums, 0.f, N_GRAPHS * 128);
    fill_f32<<<(N_GRAPHS + T - 1) / T, T, 0, stream>>>(cnt, 0.f, N_GRAPHS);
    pool_acc_kernel<<<wave_node_blocks, T, 0, stream>>>(bufA, batch, sums, cnt);
    pool_mean_kernel<<<(N_GRAPHS * 128 + T - 1) / T, T, 0, stream>>>(sums, cnt, gbuf);

    // ---- fc1 (WMMA, fused bias+ReLU), fc2 + log_softmax ----
    gemm128_wmma<<<32, T, 0, stream>>>(gbuf, fc1_w, fc1_b, h1, N_GRAPHS / 16, 1, 1);
    head_kernel<<<(N_GRAPHS + T - 1) / T, T, 0, stream>>>(h1, fc2_w, fc2_b, out);
}
